// Llama4MoE_69312182223502
// MI455X (gfx1250) — matrix operand order repaired and checked by reading, not compiled, not run on hardware
//
#include <hip/hip_runtime.h>
#include <hip/hip_bf16.h>

// ---------------------------------------------------------------------------
// Llama4-style MoE for MI455X (gfx1250, wave32, WMMA)
//   T = B*S = 2048 tokens, H = 1024, I = 2048, E = 8, top-1 routing
//
// Strategy:
//   - one-time fp32->bf16 transpose of all FFN weights into [N,K] layout so
//     WMMA B-fragments are two contiguous b128 loads per lane
//   - fused SwiGLU FFN tile kernel: 32 tokens/WG, act tile parked in LDS
//     (needs CDNA5's 320KB WGP LDS), v_wmma_f32_16x16x32_bf16 throughout
//   - x-tile staged into LDS with async global->LDS copies (ASYNCcnt path)
// ---------------------------------------------------------------------------

#define H_DIM 1024
#define I_DIM 2048
#define E_NUM 8
#define T_TOK 2048

#define BM 32            // tokens per workgroup tile
#define NWAVES 8         // 256 threads: 2 (M) x 4 (N) waves

typedef __attribute__((ext_vector_type(16))) __bf16 v16bf;
typedef __attribute__((ext_vector_type(8)))  float  v8f;

union AFrag { v16bf v; unsigned int u[8]; };
union BFrag { v16bf v; uint4 q[2]; };

static __device__ inline v8f wmma_bf16(v16bf a, v16bf b, v8f c) {
    // (neg_a, A, neg_b, B, c_mod, C, reuse_a, reuse_b)
    return __builtin_amdgcn_wmma_f32_16x16x32_bf16(false, a, false, b,
                                                   (short)0, c, false, false);
}

static __device__ inline float silu(float g) {
    return g / (1.0f + __expf(-g));
}

// async global -> LDS copy, 16 bytes per lane (tracked by ASYNCcnt)
static __device__ inline void async_copy_b128(const void* gsrc, void* lds_dst) {
    const unsigned loff = (unsigned)(size_t)lds_dst;   // generic->LDS offset (addr[31:0])
    asm volatile("global_load_async_to_lds_b128 %0, %1, off"
                 :: "v"(loff), "v"(gsrc) : "memory");
}
static __device__ inline void wait_async_lds() {
    asm volatile("s_wait_asynccnt 0" ::: "memory");
}

// ---------------------------------------------------------------------------
// 0) zero counters
// ---------------------------------------------------------------------------
__global__ void init_kernel(int* counts, int* cursor) {
    int i = threadIdx.x;
    if (i < E_NUM) { counts[i] = 0; cursor[i] = 0; }
}

// ---------------------------------------------------------------------------
// 1) weight fp32 [K,N] -> bf16 [N,K] tiled transpose (batched over experts)
// ---------------------------------------------------------------------------
__global__ __launch_bounds__(256) void transpose_convert_kernel(
        const float* __restrict__ src, __bf16* __restrict__ dst, int K, int N) {
    __shared__ float tile[32][33];
    const size_t eoff = (size_t)blockIdx.z * K * N;
    const float* s = src + eoff;
    __bf16*      d = dst + eoff;
    const int n0 = blockIdx.x * 32;
    const int k0 = blockIdx.y * 32;
    const int tx = threadIdx.x & 31;
    const int ty = threadIdx.x >> 5;   // 0..7
#pragma unroll
    for (int i = 0; i < 4; ++i)
        tile[ty + 8 * i][tx] = s[(size_t)(k0 + ty + 8 * i) * N + n0 + tx];
    __syncthreads();
#pragma unroll
    for (int i = 0; i < 4; ++i)
        d[(size_t)(n0 + ty + 8 * i) * K + k0 + tx] = (__bf16)tile[tx][ty + 8 * i];
}

// ---------------------------------------------------------------------------
// 2) router: one wave per token. logits = x @ gate_w [H,E]; top-1 + sigmoid.
// ---------------------------------------------------------------------------
__global__ __launch_bounds__(256) void router_kernel(
        const float* __restrict__ x, const float* __restrict__ gw,
        int* __restrict__ eid, float* __restrict__ score, int* __restrict__ counts) {
    const int lane = threadIdx.x & 31;
    const int wave = threadIdx.x >> 5;
    const int t = blockIdx.x * (blockDim.x >> 5) + wave;
    if (t >= T_TOK) return;

    float acc[E_NUM];
#pragma unroll
    for (int e = 0; e < E_NUM; ++e) acc[e] = 0.0f;

    for (int h0 = 0; h0 < H_DIM; h0 += 32) {
        const int h = h0 + lane;
        const float xv = x[(size_t)t * H_DIM + h];
        const float* g = &gw[(size_t)h * E_NUM];
#pragma unroll
        for (int e = 0; e < E_NUM; ++e) acc[e] += xv * g[e];
    }
#pragma unroll
    for (int e = 0; e < E_NUM; ++e)
#pragma unroll
        for (int off = 16; off > 0; off >>= 1)
            acc[e] += __shfl_xor(acc[e], off, 32);

    if (lane == 0) {
        int best = 0; float bv = acc[0];
#pragma unroll
        for (int e = 1; e < E_NUM; ++e)
            if (acc[e] > bv) { bv = acc[e]; best = e; }
        eid[t]   = best;
        score[t] = 1.0f / (1.0f + __expf(-bv));
        atomicAdd(&counts[best], 1);
    }
}

// ---------------------------------------------------------------------------
// 3) fp32 -> bf16 activations: xb = bf16(x), xsb = bf16(x*score)
// ---------------------------------------------------------------------------
__global__ __launch_bounds__(256) void convert_kernel(
        const float* __restrict__ x, const float* __restrict__ score,
        __bf16* __restrict__ xb, __bf16* __restrict__ xsb) {
    const size_t i0 = ((size_t)blockIdx.x * blockDim.x + threadIdx.x) * 4;
    if (i0 + 3 >= (size_t)T_TOK * H_DIM) return;
    const int t = (int)(i0 / H_DIM);
    const float s = score[t];
#pragma unroll
    for (int j = 0; j < 4; ++j) {
        const float v = x[i0 + j];
        xb[i0 + j]  = (__bf16)v;
        xsb[i0 + j] = (__bf16)(v * s);
    }
}

// ---------------------------------------------------------------------------
// 4) exclusive prefix over 8 expert counts
// ---------------------------------------------------------------------------
__global__ void offsets_kernel(const int* __restrict__ counts, int* __restrict__ offsets) {
    if (threadIdx.x == 0) {
        int s = 0;
#pragma unroll
        for (int e = 0; e < E_NUM; ++e) { offsets[e] = s; s += counts[e]; }
        offsets[E_NUM] = s;
    }
}

// ---------------------------------------------------------------------------
// 5) scatter tokens into per-expert buckets
// ---------------------------------------------------------------------------
__global__ __launch_bounds__(256) void scatter_kernel(
        const int* __restrict__ eid, const int* __restrict__ offsets,
        int* __restrict__ cursor, int* __restrict__ perm) {
    const int t = blockIdx.x * blockDim.x + threadIdx.x;
    if (t >= T_TOK) return;
    const int e = eid[t];
    const int p = atomicAdd(&cursor[e], 1);
    perm[offsets[e] + p] = t;
}

// ---------------------------------------------------------------------------
// 6) fused SwiGLU FFN tile kernel (shared expert or per-expert routed GEMM)
//    Weights are bf16 [N,K] (transposed), so a B-fragment = 2 x b128 per lane.
//    Phase 1: act[BM,I] = silu(X@Wg) * (X@Wu)  -> LDS (bf16)
//    Phase 2: Y[BM,H]   = act @ Wd
//    Dynamic LDS: BM*H bf16 (64KB) + BM*I bf16 (128KB) + BM ints
// ---------------------------------------------------------------------------
template <bool ROUTED>
__global__ __launch_bounds__(256, 1) void ffn_kernel(
        const __bf16* __restrict__ xb,        // [T,H] bf16 (scaled for routed)
        const int*    __restrict__ perm,      // token list (ROUTED only)
        const int*    __restrict__ offsets,   // [E+1]     (ROUTED only)
        const __bf16* __restrict__ Wg_t,      // [(E,)I,H] bf16 transposed
        const __bf16* __restrict__ Wu_t,      // [(E,)I,H]
        const __bf16* __restrict__ Wd_t,      // [(E,)H,I]
        float* __restrict__ out) {            // [T,H] fp32

    extern __shared__ char smem[];
    __bf16* lds_x   = (__bf16*)smem;                               // [BM][H]
    __bf16* lds_act = (__bf16*)(smem + (size_t)BM * H_DIM * 2);    // [BM][I]
    int*    rows    = (int*)(smem + (size_t)BM * (H_DIM + I_DIM) * 2);

    int expert = 0, tile = blockIdx.x, base = 0, cnt = T_TOK;
    if (ROUTED) {
        expert = blockIdx.x >> 6;        // 64 tiles per expert (T/BM)
        tile   = blockIdx.x & 63;
        base   = offsets[expert];
        cnt    = offsets[expert + 1] - base;
    }
    const int m0 = tile * BM;
    if (m0 >= cnt) return;               // uniform early-exit, before barriers

    const size_t estride = (size_t)H_DIM * I_DIM;
    const __bf16* Wg = Wg_t + (ROUTED ? (size_t)expert * estride : 0);
    const __bf16* Wu = Wu_t + (ROUTED ? (size_t)expert * estride : 0);
    const __bf16* Wd = Wd_t + (ROUTED ? (size_t)expert * estride : 0);

    const int tid  = threadIdx.x;
    const int lane = tid & 31;
    const int w    = tid >> 5;
    const int wm   = w >> 2;             // 0..1  -> 16-row sub-tile
    const int wn   = w & 3;              // 0..3  -> 16-col sub-tile stride 64
    const int mb   = wm * 16;

    // token row indices for this tile (clamped for ragged tail)
    if (tid < BM) {
        int r = m0 + tid;
        int rc = (r < cnt) ? r : (cnt - 1);
        rows[tid] = ROUTED ? perm[base + rc] : rc;
    }
    __syncthreads();

    // stage X tile into LDS with async global->LDS b128 copies
    {
        const int r = tid >> 3;          // 0..31
        const int c = tid & 7;           // 8 chunks of 128 bf16
        const __bf16* src = xb + (size_t)rows[r] * H_DIM + c * 128;
        __bf16* dst = lds_x + (size_t)r * H_DIM + c * 128;
#pragma unroll
        for (int i = 0; i < 16; ++i)
            async_copy_b128(src + i * 8, dst + i * 8);
        wait_async_lds();                // this wave's copies landed
    }
    __syncthreads();                     // everyone's copies landed

    const int acol  = lane & 15;         // A/B fragment column / C-D N index
    const int ahalf = lane >> 4;         // lane group 0/1
    const int kbA   = ahalf * 16;        // A frag: K base (K=0..15 / 16..31)
    const int kbB   = ahalf * 16;        // B frag: K base (K=0..15 / 16..31)

    // ----------------- Phase 1: gate/up + SiLU -> lds_act -----------------
    for (int n0 = wn * 16; n0 < I_DIM; n0 += 64) {
        v8f accg = {};
        v8f accu = {};
        const int ncol = n0 + acol;
        const __bf16* rowg = Wg + (size_t)ncol * H_DIM + kbB;   // walk along K
        const __bf16* rowu = Wu + (size_t)ncol * H_DIM + kbB;
        for (int k0 = 0; k0 < H_DIM; k0 += 32) {
            // A fragment: 16x32 bf16 from lds_x rows [mb, mb+16)
            AFrag af;
            const __bf16* rowp = lds_x + (size_t)(mb + acol) * H_DIM + k0;
#pragma unroll
            for (int v = 0; v < 4; ++v) {
                af.u[v]     = *(const unsigned int*)(rowp + kbA + 2 * v);
                af.u[4 + v] = *(const unsigned int*)(rowp + 8 + kbA + 2 * v);
            }
            // B fragments: 16 contiguous bf16 along K -> 2 x b128
            BFrag bg, bu;
            const uint4* pg = (const uint4*)(rowg + k0);
            const uint4* pu = (const uint4*)(rowu + k0);
            bg.q[0] = pg[0]; bg.q[1] = pg[1];
            bu.q[0] = pu[0]; bu.q[1] = pu[1];
            __builtin_prefetch(pg + 4, 0, 1);   // next K-step (32 elem = 4 uint4)
            __builtin_prefetch(pu + 4, 0, 1);
            accg = wmma_bf16(af.v, bg.v, accg);
            accu = wmma_bf16(af.v, bu.v, accu);
        }
        // D layout: VGPR r -> M = mb + r (+8 for lanes 16..31), N = n0 + (lane&15)
#pragma unroll
        for (int r = 0; r < 8; ++r) {
            const int m = mb + r + ahalf * 8;
            const float a = silu(accg[r]) * accu[r];
            lds_act[(size_t)m * I_DIM + ncol] = (__bf16)a;
        }
    }
    __syncthreads();

    // ----------------- Phase 2: down projection ---------------------------
    for (int n0 = wn * 16; n0 < H_DIM; n0 += 64) {
        v8f acc = {};
        const int ncol = n0 + acol;
        const __bf16* rowd = Wd + (size_t)ncol * I_DIM + kbB;
        for (int k0 = 0; k0 < I_DIM; k0 += 32) {
            AFrag af;
            const __bf16* rowp = lds_act + (size_t)(mb + acol) * I_DIM + k0;
#pragma unroll
            for (int v = 0; v < 4; ++v) {
                af.u[v]     = *(const unsigned int*)(rowp + kbA + 2 * v);
                af.u[4 + v] = *(const unsigned int*)(rowp + 8 + kbA + 2 * v);
            }
            BFrag bd;
            const uint4* pd = (const uint4*)(rowd + k0);
            bd.q[0] = pd[0]; bd.q[1] = pd[1];
            __builtin_prefetch(pd + 4, 0, 1);
            acc = wmma_bf16(af.v, bd.v, acc);
        }
#pragma unroll
        for (int r = 0; r < 8; ++r) {
            const int m = mb + r + ahalf * 8;
            if (m0 + m < cnt) {
                const int tok = rows[m];
                float* dst = &out[(size_t)tok * H_DIM + ncol];
                if (ROUTED) atomicAdd(dst, acc[r]);
                else        *dst = acc[r];
            }
        }
    }
}

// ---------------------------------------------------------------------------
// launch
// ---------------------------------------------------------------------------
extern "C" void kernel_launch(void* const* d_in, const int* in_sizes, int n_in,
                              void* d_out, int out_size, void* d_ws, size_t ws_size,
                              hipStream_t stream) {
    const float* x       = (const float*)d_in[0];  // [T,H]
    const float* gate_w  = (const float*)d_in[1];  // [H,E]
    const float* sh_gate = (const float*)d_in[2];  // [H,I]
    const float* sh_up   = (const float*)d_in[3];  // [H,I]
    const float* sh_down = (const float*)d_in[4];  // [I,H]
    const float* rt_gate = (const float*)d_in[5];  // [E,H,I]
    const float* rt_up   = (const float*)d_in[6];  // [E,H,I]
    const float* rt_down = (const float*)d_in[7];  // [E,I,H]
    float* out = (float*)d_out;                    // [T,H]

    const size_t HI = (size_t)H_DIM * I_DIM;

    // workspace layout (transposed bf16 weights ~108MB + activations ~8.3MB)
    char* ws = (char*)d_ws;
    __bf16* shg_t = (__bf16*)ws;  ws += HI * 2;            // [I,H]
    __bf16* shu_t = (__bf16*)ws;  ws += HI * 2;            // [I,H]
    __bf16* shd_t = (__bf16*)ws;  ws += HI * 2;            // [H,I]
    __bf16* rtg_t = (__bf16*)ws;  ws += E_NUM * HI * 2;    // [E,I,H]
    __bf16* rtu_t = (__bf16*)ws;  ws += E_NUM * HI * 2;    // [E,I,H]
    __bf16* rtd_t = (__bf16*)ws;  ws += E_NUM * HI * 2;    // [E,H,I]
    __bf16* xb    = (__bf16*)ws;  ws += (size_t)T_TOK * H_DIM * 2;
    __bf16* xsb   = (__bf16*)ws;  ws += (size_t)T_TOK * H_DIM * 2;
    float* score  = (float*)ws;   ws += (size_t)T_TOK * 4;
    int* eid      = (int*)ws;     ws += (size_t)T_TOK * 4;
    int* perm     = (int*)ws;     ws += (size_t)T_TOK * 4;
    int* counts   = (int*)ws;     ws += 64;
    int* offsets  = (int*)ws;     ws += 64;
    int* cursor   = (int*)ws;     ws += 64;

    const size_t smem = (size_t)BM * (H_DIM + I_DIM) * 2 + BM * sizeof(int);

    // one-time per call: weight transpose + bf16 conversion
    transpose_convert_kernel<<<dim3(I_DIM / 32, H_DIM / 32, 1), 256, 0, stream>>>(sh_gate, shg_t, H_DIM, I_DIM);
    transpose_convert_kernel<<<dim3(I_DIM / 32, H_DIM / 32, 1), 256, 0, stream>>>(sh_up,   shu_t, H_DIM, I_DIM);
    transpose_convert_kernel<<<dim3(H_DIM / 32, I_DIM / 32, 1), 256, 0, stream>>>(sh_down, shd_t, I_DIM, H_DIM);
    transpose_convert_kernel<<<dim3(I_DIM / 32, H_DIM / 32, E_NUM), 256, 0, stream>>>(rt_gate, rtg_t, H_DIM, I_DIM);
    transpose_convert_kernel<<<dim3(I_DIM / 32, H_DIM / 32, E_NUM), 256, 0, stream>>>(rt_up,   rtu_t, H_DIM, I_DIM);
    transpose_convert_kernel<<<dim3(H_DIM / 32, I_DIM / 32, E_NUM), 256, 0, stream>>>(rt_down, rtd_t, I_DIM, H_DIM);

    init_kernel<<<1, 64, 0, stream>>>(counts, cursor);
    router_kernel<<<T_TOK / 8, 256, 0, stream>>>(x, gate_w, eid, score, counts);
    convert_kernel<<<(T_TOK * H_DIM) / (256 * 4), 256, 0, stream>>>(x, score, xb, xsb);
    offsets_kernel<<<1, 32, 0, stream>>>(counts, offsets);
    scatter_kernel<<<T_TOK / 256, 256, 0, stream>>>(eid, offsets, cursor, perm);

    // shared expert: writes out directly
    ffn_kernel<false><<<T_TOK / BM, 256, smem, stream>>>(
        xb, nullptr, nullptr, shg_t, shu_t, shd_t, out);
    // routed experts: worst-case grid, device-side early exit; f32 atomic adds
    ffn_kernel<true><<<E_NUM * (T_TOK / BM), 256, smem, stream>>>(
        xsb, perm, offsets, rtg_t, rtu_t, rtd_t, out);
}
